// GraphAttention_64381559767431
// MI455X (gfx1250) — matrix operand order repaired and checked by reading, not compiled, not run on hardware
//
#include <hip/hip_runtime.h>
#include <hip/hip_bf16.h>
#include <math.h>

#define BB 4
#define NN 2048
#define FF 64
#define HH 4
#define KK 64
#define TROW 36   // padded LDS row stride (floats); 144B, 16B aligned

typedef __attribute__((ext_vector_type(16))) __bf16 v16bf;
typedef __attribute__((ext_vector_type(2)))  __bf16 bf16x2;
typedef __attribute__((ext_vector_type(8)))  float  v8f;
typedef __attribute__((ext_vector_type(4)))  unsigned u32x4;
typedef __attribute__((ext_vector_type(8)))  unsigned u32x8;

union BFrag {
  v16bf v;
  unsigned u[8];
  uint4 q[2];
};

// pack two f32 into bf16x2 (low = a, high = b), RNE
__device__ __forceinline__ unsigned pack2_bf16(float a, float b) {
#if __has_builtin(__builtin_amdgcn_cvt_pk_bf16_f32)
  union { bf16x2 v; unsigned u; } r;
  r.v = __builtin_amdgcn_cvt_pk_bf16_f32(a, b);
  return r.u;
#else
  union { float f; unsigned u; } ua, ub;
  ua.f = a; ub.f = b;
  unsigned x = ua.u + (0x7FFFu + ((ua.u >> 16) & 1u));
  unsigned y = ub.u + (0x7FFFu + ((ub.u >> 16) & 1u));
  return (x >> 16) | (y & 0xFFFF0000u);
#endif
}

__global__ void init_kernel(float* p, int n) {
  int i = threadIdx.x + blockIdx.x * blockDim.x;
  if (i < n) p[i] = 0.0f;
}

// ---------------------------------------------------------------------------
// Kernel A: feats[b,h] = x[b] @ W[h]  via v_wmma_f32_16x16x32_bf16.
// ---------------------------------------------------------------------------
__global__ __launch_bounds__(128) void feats_kernel(
    const float* __restrict__ x, const float* __restrict__ W,
    const float* __restrict__ a_self, const float* __restrict__ a_neigh,
    unsigned short* __restrict__ featsT,
    float* __restrict__ sSelf, float* __restrict__ sNeigh)
{
  const int lane = threadIdx.x & 31;
  const int wid  = threadIdx.x >> 5;
  const int ntiles = NN / 16;
  int tile = blockIdx.x * 4 + wid;
  int bh = tile / ntiles;
  int it = tile % ntiles;
  int b = bh / HH, h = bh % HH;
  int i0 = it * 16;

  const int half = (lane >= 16) ? 1 : 0;
  const int lm   = lane & 15;

  v8f zero = {};
  v8f acc[4];
#pragma unroll
  for (int t = 0; t < 4; ++t) acc[t] = zero;

  const float* xrow = x + ((size_t)b * NN + i0) * FF;
  const float* Wh   = W + (size_t)h * FF * KK;

  for (int f0 = 0; f0 < FF; f0 += 32) {
    BFrag A;
    {
      const float* xr = xrow + (size_t)lm * FF;
      int kbase = f0 + (half ? 16 : 0);
#pragma unroll
      for (int v = 0; v < 4; ++v) {
        float2 p = *(const float2*)(xr + kbase + 2 * v);
        A.u[v] = pack2_bf16(p.x, p.y);
      }
#pragma unroll
      for (int v = 0; v < 4; ++v) {
        float2 p = *(const float2*)(xr + kbase + 8 + 2 * v);
        A.u[4 + v] = pack2_bf16(p.x, p.y);
      }
    }
    int rbase = f0 + (half ? 16 : 0);
#pragma unroll
    for (int kt = 0; kt < 4; ++kt) {
      BFrag Bf;
      int col = kt * 16 + lm;
#pragma unroll
      for (int v = 0; v < 8; ++v) {
        float e0 = Wh[(size_t)(rbase + 2 * v) * KK + col];
        float e1 = Wh[(size_t)(rbase + 2 * v + 1) * KK + col];
        Bf.u[v] = pack2_bf16(e0, e1);
      }
      acc[kt] = __builtin_amdgcn_wmma_f32_16x16x32_bf16(
          false, A.v, false, Bf.v, (short)0, acc[kt], false, false);
    }
  }

  float accS[8], accN[8];
#pragma unroll
  for (int r = 0; r < 8; ++r) { accS[r] = 0.f; accN[r] = 0.f; }

#pragma unroll
  for (int kt = 0; kt < 4; ++kt) {
    int k = kt * 16 + lm;
    float as = a_self[h * KK + k];
    float an = a_neigh[h * KK + k];
    size_t tbase = ((size_t)bh * KK + k) * NN + i0 + (half ? 8 : 0);
#pragma unroll
    for (int r = 0; r < 8; ++r) {
      float val = acc[kt][r];
      featsT[tbase + r] = (unsigned short)pack2_bf16(val, 0.f);
      accS[r] += val * as;
      accN[r] += val * an;
    }
  }
#pragma unroll
  for (int r = 0; r < 8; ++r) {
#pragma unroll
    for (int off = 1; off < 16; off <<= 1) {
      accS[r] += __shfl_xor(accS[r], off, 32);
      accN[r] += __shfl_xor(accN[r], off, 32);
    }
  }
  if (lm == 0) {
    size_t mbase = (size_t)bh * NN + i0 + (half ? 8 : 0);
#pragma unroll
    for (int r = 0; r < 8; ++r) {
      sSelf[mbase + r]  = accS[r];
      sNeigh[mbase + r] = accN[r];
    }
  }
}

// ---------------------------------------------------------------------------
// Kernel B: per-row softmax stats (max, 1/l) + aux losses.
// ---------------------------------------------------------------------------
__global__ __launch_bounds__(256) void stats_kernel(
    const float* __restrict__ adj, const float* __restrict__ mask,
    const float* __restrict__ sSelf, const float* __restrict__ sNeigh,
    float* __restrict__ rowMax, float* __restrict__ rowInvL,
    float* __restrict__ uloss, float* __restrict__ eloss)
{
  const int lane = threadIdx.x & 31;
  const int wid  = threadIdx.x >> 5;
  size_t row = (size_t)blockIdx.x * 8 + wid;
  int b = (int)(row / ((size_t)HH * NN));
  int i = (int)(row % NN);

  const float si = sSelf[row];
  const float* sN   = sNeigh + (row / NN) * NN;
  const float* mrow = mask + ((size_t)b * NN + i) * NN;
  const float* arow = adj  + ((size_t)b * NN + i) * NN;

  float mx = -3.4e38f;
  for (int j = lane; j < NN; j += 32) {
    float e = si + sN[j];
    e = (e > 0.f) ? e : 0.2f * e;
    e += mrow[j];
    mx = fmaxf(mx, e);
  }
  for (int off = 1; off < 32; off <<= 1) mx = fmaxf(mx, __shfl_xor(mx, off, 32));

  float l = 0.f, s2 = 0.f, cnt = 0.f, deg = 0.f;
  for (int j = lane; j < NN; j += 32) {
    float e = si + sN[j];
    e = (e > 0.f) ? e : 0.2f * e;
    e += mrow[j];
    float ex = __expf(e - mx);
    float a  = arow[j];
    float p  = ex * a;
    l  += ex;
    s2 += p;
    cnt += (p != 0.f) ? 1.f : 0.f;
    deg += (a != 0.f) ? 1.f : 0.f;
  }
  for (int off = 1; off < 32; off <<= 1) {
    l   += __shfl_xor(l,   off, 32);
    s2  += __shfl_xor(s2,  off, 32);
    cnt += __shfl_xor(cnt, off, 32);
    deg += __shfl_xor(deg, off, 32);
  }
  if (lane == 0) {
    float invl = 1.f / l;
    rowMax[row]  = mx;
    rowInvL[row] = invl;
    const float invN = 1.0f / (float)NN;
    atomicAdd(uloss + b, (cnt - deg) * invN);
    atomicAdd(eloss + b, s2 * invl * invN);
  }
}

// ---------------------------------------------------------------------------
// Kernel C: out = softmax*adj @ feats.
// mask tiles: TDM tensor_load_to_lds (2D D#, HW row padding to TROW).
// adj  tiles: global_load_async_to_lds_b128.
// Both double-buffered; s_wait_tensorcnt / s_wait_asynccnt gate consumption.
// ---------------------------------------------------------------------------

// TDM 2D tile load: 16 rows x 32 f32, row stride NN, LDS rows padded to TROW.
__device__ __forceinline__ void tdm_load_tile(unsigned long long gaddr,
                                              unsigned lds_addr,
                                              const u32x8& g1)
{
  u32x4 g0;
  g0.x = 1u;                                            // count=1, user desc
  g0.y = lds_addr;                                      // LDS byte address
  g0.z = (unsigned)gaddr;                               // global_addr[31:0]
  g0.w = (unsigned)((gaddr >> 32) & 0x01FFFFFFull)      // global_addr[56:32]
       | (2u << 30);                                    // type = 2 ("image")
  asm volatile("tensor_load_to_lds %0, %1" :: "s"(g0), "s"(g1) : "memory");
}

__device__ __forceinline__ void issue_adj(const float* arowBase,
                                          int j0, int lane, unsigned ldsBase)
{
#pragma unroll
  for (int t = 0; t < 4; ++t) {
    int s = t * 32 + lane;          // 128 segments of 16B
    int row = s >> 3;
    int jo  = (s & 7) * 4;
    const float* ga = arowBase + (size_t)row * NN + j0 + jo;
    unsigned la_off = ldsBase + (unsigned)(TROW * 16 * 4) +
                      (unsigned)((row * TROW + jo) * 4);
    asm volatile("global_load_async_to_lds_b128 %0, %1, off"
                 :: "v"(la_off), "v"((unsigned long long)(size_t)ga) : "memory");
  }
}

__global__ __launch_bounds__(128) void out_kernel(
    const float* __restrict__ adj, const float* __restrict__ mask,
    const unsigned short* __restrict__ featsT,
    const float* __restrict__ sSelf, const float* __restrict__ sNeigh,
    const float* __restrict__ rowMax, const float* __restrict__ rowInvL,
    const float* __restrict__ bias, float* __restrict__ act)
{
  __shared__ __align__(16) float sNl[NN];                    // 8 KB
  __shared__ __align__(16) float tiles[4][2][2 * 16 * TROW]; // 36 KB dbl-buf

  const int lane = threadIdx.x & 31;
  const int wid  = threadIdx.x >> 5;
  const int ntiles = NN / 16;
  int tile = blockIdx.x * 4 + wid;
  int bh = tile / ntiles;       // uniform across block (128 % 4 == 0)
  int it = tile % ntiles;
  int b = bh / HH, h = bh % HH;
  int i0 = it * 16;
  const int half = (lane >= 16) ? 1 : 0;
  const int lm   = lane & 15;

  {
    const float* sN = sNeigh + (size_t)bh * NN;
    for (int t = threadIdx.x; t < NN / 4; t += 128)
      ((float4*)sNl)[t] = ((const float4*)sN)[t];
  }
  __syncthreads();

  int irow = i0 + lm;
  size_t rbase = (size_t)bh * NN + irow;
  float si   = sSelf[rbase];
  float mi   = rowMax[rbase];
  float invl = rowInvL[rbase];
  const float* mrowBase = mask + ((size_t)b * NN + i0) * NN;
  const float* arowBase = adj  + ((size_t)b * NN + i0) * NN;
  const unsigned short* ftb = featsT + (size_t)bh * KK * NN;

  unsigned ldsBase0 = (unsigned)(size_t)&tiles[wid][0][0];
  unsigned ldsBase1 = (unsigned)(size_t)&tiles[wid][1][0];

  // D# group1 (shared by all mask-tile loads):
  //  w0: data_size=4B (2<<16), pad_enable (1<<20), pad_interval=32dw (4<<22),
  //      pad_amount=4dw (3<<25)
  //  w1..w3: tensor_dim0=2048, tensor_dim1=2048, tile_dim0=32
  //  w4: tile_dim1=16; w5: tensor_dim0_stride=2048
  const u32x8 g1 = { 0x07120000u, 0x08000000u, 0x08000000u, 0x00200000u,
                     16u, 2048u, 0u, 0u };

  v8f zero = {};
  v8f acc[4];
#pragma unroll
  for (int t = 0; t < 4; ++t) acc[t] = zero;

  tdm_load_tile((unsigned long long)(size_t)(mrowBase + 0), ldsBase0, g1);
  issue_adj(arowBase, 0, lane, ldsBase0);

  for (int j0 = 0; j0 < NN; j0 += 32) {
    int cur = (j0 >> 5) & 1;
    if (j0 + 32 < NN) {
      unsigned nextBase = cur ? ldsBase0 : ldsBase1;
      tdm_load_tile((unsigned long long)(size_t)(mrowBase + j0 + 32), nextBase, g1);
      issue_adj(arowBase, j0 + 32, lane, nextBase);
      __builtin_prefetch(arowBase + j0 + 64, 0, 0);        // warm L2 ahead
      asm volatile("s_wait_tensorcnt 0x1" ::: "memory");   // cur mask tile done
      asm volatile("s_wait_asynccnt 0x4" ::: "memory");    // cur adj tile done
    } else {
      asm volatile("s_wait_tensorcnt 0x0" ::: "memory");
      asm volatile("s_wait_asynccnt 0x0" ::: "memory");
    }

    const float* mT = &tiles[wid][cur][0];
    const float* aT = &tiles[wid][cur][16 * TROW];

    // A fragment: P[i0..i0+15, j0..j0+31] in bf16, built from LDS
    BFrag A;
    int jb = half ? 16 : 0;
#pragma unroll
    for (int v = 0; v < 8; ++v) {
      int j = jb + 2 * v;
      float2 sn = *(const float2*)(sNl + j0 + j);
      float2 mk = *(const float2*)(mT + lm * TROW + j);
      float2 aj = *(const float2*)(aT + lm * TROW + j);
      float e0 = si + sn.x; e0 = (e0 > 0.f) ? e0 : 0.2f * e0; e0 += mk.x;
      float e1 = si + sn.y; e1 = (e1 > 0.f) ? e1 : 0.2f * e1; e1 += mk.y;
      float p0 = __expf(e0 - mi) * invl * aj.x;
      float p1 = __expf(e1 - mi) * invl * aj.y;
      A.u[v] = pack2_bf16(p0, p1);
    }
    // B fragments: featsT rows contiguous in j -> b128 loads
    int jB = j0 + (half ? 16 : 0);
#pragma unroll
    for (int kt = 0; kt < 4; ++kt) {
      BFrag Bf;
      const unsigned short* fp = ftb + (size_t)(kt * 16 + lm) * NN + jB;
      Bf.q[0] = *(const uint4*)(fp);
      Bf.q[1] = *(const uint4*)(fp + 8);
      acc[kt] = __builtin_amdgcn_wmma_f32_16x16x32_bf16(
          false, A.v, false, Bf.v, (short)0, acc[kt], false, false);
    }
  }

  const float BN_SCALE = 0.9995003746877732f;   // 1/sqrt(1+1e-3)
#pragma unroll
  for (int kt = 0; kt < 4; ++kt) {
    int k = kt * 16 + lm;
    float bv = bias[h * KK + k];
#pragma unroll
    for (int r = 0; r < 8; ++r) {
      int m = r + (half ? 8 : 0);
      float v = (acc[kt][r] + bv) * BN_SCALE;
      v = fmaxf(v, 0.f);
      act[(((size_t)b * NN) + i0 + m) * (HH * KK) + h * KK + k] = v;
    }
  }
}

// ---------------------------------------------------------------------------
extern "C" void kernel_launch(void* const* d_in, const int* in_sizes, int n_in,
                              void* d_out, int out_size, void* d_ws, size_t ws_size,
                              hipStream_t stream) {
  (void)in_sizes; (void)n_in; (void)out_size; (void)ws_size;
  const float* x       = (const float*)d_in[0];
  const float* adj     = (const float*)d_in[1];
  const float* mask    = (const float*)d_in[2];
  const float* W       = (const float*)d_in[3];
  const float* a_self  = (const float*)d_in[4];
  const float* a_neigh = (const float*)d_in[5];
  const float* bias    = (const float*)d_in[6];

  float* act   = (float*)d_out;
  float* uloss = act + (size_t)BB * NN * HH * KK;

  char* ws = (char*)d_ws;
  unsigned short* featsT = (unsigned short*)ws;                 // 4 MB bf16
  size_t featsBytes = (size_t)BB * HH * KK * NN * sizeof(unsigned short);
  float* sSelf   = (float*)(ws + featsBytes);
  float* sNeigh  = sSelf  + (size_t)BB * HH * NN;
  float* rowMax  = sNeigh + (size_t)BB * HH * NN;
  float* rowInvL = rowMax + (size_t)BB * HH * NN;
  float* eloss   = uloss + BB;

  init_kernel<<<1, 32, 0, stream>>>(uloss, 2 * BB);
  feats_kernel<<<BB * HH * (NN / 16) / 4, 128, 0, stream>>>(
      x, W, a_self, a_neigh, featsT, sSelf, sNeigh);
  stats_kernel<<<BB * HH * NN / 8, 256, 0, stream>>>(
      adj, mask, sSelf, sNeigh, rowMax, rowInvL, uloss, eloss);
  out_kernel<<<BB * HH * (NN / 16) / 4, 128, 0, stream>>>(
      adj, mask, featsT, sSelf, sNeigh, rowMax, rowInvL, bias, act);
}
